// SelfAttention_36369783062745
// MI455X (gfx1250) — matrix operand order repaired and checked by reading, not compiled, not run on hardware
//
#include <hip/hip_runtime.h>
#include <hip/hip_fp16.h>

// ---------------------------------------------------------------------------
// Self-attention block (GroupNorm -> qkv 1x1conv -> MHA -> proj -> residual)
// B=16, C=256, N=HW=1024, heads=4, head_dim=64, groups=8.
// WMMA f16 (f32 accum) GEMMs with 32x64 register-blocked wave tiles;
// flash attention with TDM (tensor_load_to_lds) double-buffered K/V staging.
// Layouts chosen so every WMMA fragment is a contiguous 16B load:
//   h_t   [B][s][256]   (seq-major)      -> qkv GEMM B-frags contiguous
//   qk_t  [B][s][512]   (Q|K seq-major)  -> Q A-frags, K B-frags contiguous
//   v_buf [B][256][1024] (d-major)       -> PV B-frags contiguous ([d][j] tile)
//   attn_t[B][s][256]   (seq-major)      -> proj GEMM B-frags contiguous
// ---------------------------------------------------------------------------

typedef __attribute__((ext_vector_type(16))) _Float16 v16h;
typedef __attribute__((ext_vector_type(8)))  _Float16 v8h;
typedef __attribute__((ext_vector_type(8)))  float    v8f;
typedef __attribute__((ext_vector_type(4)))  unsigned int v4u;
typedef __attribute__((ext_vector_type(8)))  unsigned int v8u;

constexpr int kB  = 16;
constexpr int kC  = 256;
constexpr int kN  = 1024;
constexpr int kG  = 8;
constexpr int kHD = 64;

// ---------------------------------------------------------------------------
// WMMA fragment helpers (ISA 7.12.2, wave32)
//  A 16x32: lane row = lane&15; K-half (0/16) by lane>>4; contiguous K (this code's order, not the ISA table's).
//  B 32x16: lane col = lane&15; K-half (0/16) by lane>>4, same code order as A; contiguous K when
//           storage is [n][k].
//  C/D 16x16 f32: row = r + 8*(lane>>4), col = lane&15.
// ---------------------------------------------------------------------------
__device__ inline v16h load_a_contig(const _Float16* ap, int hi) {
  const int kh = hi * 16;
  v8h lo = *reinterpret_cast<const v8h*>(ap + kh);
  v8h up = *reinterpret_cast<const v8h*>(ap + kh + 8);
  v16h a;
  #pragma unroll
  for (int j = 0; j < 8; ++j) { a[j] = lo[j]; a[8 + j] = up[j]; }
  return a;
}

__device__ inline v16h load_b_contig(const _Float16* bp) {
  // bp points at storage[n][k0 + hi*16]; 16 contiguous k
  v8h lo = *reinterpret_cast<const v8h*>(bp);
  v8h up = *reinterpret_cast<const v8h*>(bp + 8);
  v16h b;
  #pragma unroll
  for (int j = 0; j < 8; ++j) { b[j] = lo[j]; b[8 + j] = up[j]; }
  return b;
}

// ---------------------------------------------------------------------------
// Tensor Data Mover: 2-D tile (64x64, 16-bit elems) global -> LDS with row
// padding so the tile lands as [64][72].  D# per ISA 8.3/8.4.
// ---------------------------------------------------------------------------
__device__ inline void tdm_load_tile_64x64(unsigned lds_addr, const _Float16* gsrc,
                                           unsigned row_stride_elems,
                                           unsigned tensor_rows) {
  unsigned long long ga = (unsigned long long)(const void*)gsrc;
  v4u g0;
  g0[0] = 1u;                                            // count=1, no gather
  g0[1] = lds_addr;                                      // LDS byte address
  g0[2] = (unsigned)(ga & 0xffffffffu);                  // global_addr[31:0]
  g0[3] = (unsigned)((ga >> 32) & 0x01ffffffu) | (2u << 30);  // addr[56:32] | type=2
  v8u g1;
  // data_size=2B (1<<16), pad_enable (1<<20), pad_interval=4 (32 DW), pad_amount=3 (4 DW)
  g1[0] = (1u << 16) | (1u << 20) | (4u << 22) | (3u << 25);
  const unsigned td0 = row_stride_elems;                 // tensor_dim0
  const unsigned td1 = tensor_rows;                      // tensor_dim1
  g1[1] = (td0 & 0xffffu) << 16;                         // tensor_dim0[15:0] @ bits63:48
  g1[2] = (td0 >> 16) | ((td1 & 0xffffu) << 16);         // dim0 hi | dim1 lo
  g1[3] = (td1 >> 16) | (64u << 16);                     // dim1 hi | tile_dim0=64
  g1[4] = 64u;                                           // tile_dim1=64, tile_dim2=0
  g1[5] = row_stride_elems;                              // tensor_dim0_stride lo32
  g1[6] = 0u;
  g1[7] = 0u;
  asm volatile("tensor_load_to_lds %0, %1" :: "s"(g0), "s"(g1) : "memory");
}

// ---------------------------------------------------------------------------
// Kernel 1: f32 -> f16 weight convert; first `scaleRows` rows scaled.
// ---------------------------------------------------------------------------
__global__ void cvt_w_kernel(const float* __restrict__ src, _Float16* __restrict__ dst,
                             int n, int scaleRows, int ld, float scale) {
  int i = blockIdx.x * blockDim.x + threadIdx.x;
  if (i < n) {
    float v = src[i];
    if (i / ld < scaleRows) v *= scale;
    dst[i] = (_Float16)v;
  }
}

// ---------------------------------------------------------------------------
// Kernel 2: GroupNorm -> h_t [B][s][256] f16 (seq-major, contiguous writes)
// ---------------------------------------------------------------------------
__global__ void groupnorm_kernel(const float* __restrict__ x,
                                 const float* __restrict__ w,
                                 const float* __restrict__ bb,
                                 _Float16* __restrict__ ht) {
  const int bg = blockIdx.x;
  const int batch = bg / kG, g = bg % kG;
  const int cpg = kC / kG;                 // 32
  const int n   = cpg * kN;                // 32768
  const float* xg = x + (size_t)batch * kC * kN + (size_t)g * cpg * kN;

  float s = 0.f, ss = 0.f;
  for (int i = threadIdx.x; i < n; i += blockDim.x) {
    float v = xg[i];
    s += v; ss += v * v;
  }
  __shared__ float red[2][8];
  #pragma unroll
  for (int off = 16; off >= 1; off >>= 1) {
    s  += __shfl_xor(s,  off, 32);
    ss += __shfl_xor(ss, off, 32);
  }
  int wid = threadIdx.x >> 5, lane = threadIdx.x & 31;
  if (lane == 0) { red[0][wid] = s; red[1][wid] = ss; }
  __syncthreads();
  if (wid == 0) {
    s  = (lane < 8) ? red[0][lane] : 0.f;
    ss = (lane < 8) ? red[1][lane] : 0.f;
    #pragma unroll
    for (int off = 4; off >= 1; off >>= 1) {
      s  += __shfl_xor(s,  off, 32);
      ss += __shfl_xor(ss, off, 32);
    }
    if (lane == 0) { red[0][0] = s; red[1][0] = ss; }
  }
  __syncthreads();
  float mean = red[0][0] / (float)n;
  float var  = red[1][0] / (float)n - mean * mean;
  float rstd = rsqrtf(var + 1e-5f);

  // pass 2: strided reads (x is L2-hot), contiguous f16 writes [s][c]
  _Float16* hb = ht + (size_t)batch * kN * kC + g * cpg;
  for (int i = threadIdx.x; i < n; i += blockDim.x) {
    int cl = i & 31, sp = i >> 5;
    float v = (xg[(size_t)cl * kN + sp] - mean) * rstd * w[g * cpg + cl] + bb[g * cpg + cl];
    hb[(size_t)sp * kC + cl] = (_Float16)v;
  }
}

// ---------------------------------------------------------------------------
// Kernel 3: QKV GEMM.  acc[o][s] = sum_c Wq[o][c] * h_t[s][c]
//   o <  512 (Q|K): store transposed -> qk_t[B][s][512] (v8h packed stores)
//   o >= 512 (V)  : store d-major    -> v_buf[B][256][1024]
// Block 256 thr = 8 waves (4x2); block tile 128x128; wave tile 32(M) x 64(N)
// with 2 A-frags reusing each B-frag (8 WMMAs per k-step per wave).
// ---------------------------------------------------------------------------
__global__ void gemm_qkv_kernel(const _Float16* __restrict__ W,    // [768,256]
                                const _Float16* __restrict__ Ht,   // [B,1024,256]
                                _Float16* __restrict__ QKt,        // [B,1024,512]
                                _Float16* __restrict__ Vb) {       // [B,256,1024]
  constexpr int K = 256;
  const int b  = blockIdx.z;
  const int m0 = blockIdx.x * 128;
  const int n0 = blockIdx.y * 128;
  const int wid = threadIdx.x >> 5, lane = threadIdx.x & 31;
  const int mw = m0 + (wid >> 1) * 32;
  const int nw = n0 + (wid & 1) * 64;
  const int row = lane & 15, hi = lane >> 4;
  const _Float16* Bp = Ht + (size_t)b * kN * kC;

  v8f acc[2][4] = {};
  for (int k0 = 0; k0 < K; k0 += 32) {
    v16h a0 = load_a_contig(W + (size_t)(mw + row) * K + k0, hi);
    v16h a1 = load_a_contig(W + (size_t)(mw + 16 + row) * K + k0, hi);
    #pragma unroll
    for (int t = 0; t < 4; ++t) {
      v16h bf = load_b_contig(Bp + (size_t)(nw + t * 16 + row) * kC + k0 + hi * 16);
      acc[0][t] = __builtin_amdgcn_wmma_f32_16x16x32_f16(false, a0, false, bf,
                                                         (short)0, acc[0][t], false, false);
      acc[1][t] = __builtin_amdgcn_wmma_f32_16x16x32_f16(false, a1, false, bf,
                                                         (short)0, acc[1][t], false, false);
    }
  }
  if (mw < 512) {
    _Float16* op = QKt + (size_t)b * kN * 512;
    #pragma unroll
    for (int mi = 0; mi < 2; ++mi)
      #pragma unroll
      for (int t = 0; t < 4; ++t) {
        const int n = nw + t * 16 + row;
        v8h h8;
        #pragma unroll
        for (int r = 0; r < 8; ++r) h8[r] = (_Float16)acc[mi][t][r];
        *reinterpret_cast<v8h*>(&op[(size_t)n * 512 + mw + mi * 16 + 8 * hi]) = h8;
      }
  } else {
    _Float16* op = Vb + (size_t)b * kC * kN + (size_t)(mw - 512) * kN;
    #pragma unroll
    for (int mi = 0; mi < 2; ++mi)
      #pragma unroll
      for (int t = 0; t < 4; ++t)
        #pragma unroll
        for (int r = 0; r < 8; ++r)
          op[(size_t)(mi * 16 + r + 8 * hi) * kN + nw + t * 16 + row] = (_Float16)acc[mi][t][r];
  }
}

// ---------------------------------------------------------------------------
// Kernel 4: flash attention per (b,h).  Q pre-scaled (0.125 folded into Wq).
// Block = 128 thr (4 waves), 64 query rows.  j-loop in 64-col chunks; K and V
// tiles DMA'd by the TDM into double-buffered padded LDS ([64][72]); wave 0
// issues/waits TENSORcnt, everyone else just computes.  16 WMMAs/chunk/wave.
// ---------------------------------------------------------------------------
__global__ void attn_kernel(const _Float16* __restrict__ QKt,  // [B,1024,512]
                            const _Float16* __restrict__ Vb,   // [B,256,1024]
                            _Float16* __restrict__ At) {       // [B,1024,256]
  constexpr int JC = 64, NCHUNK = kN / JC;
  const int bh  = blockIdx.y;              // 0..63
  const int b   = bh >> 2, hh = bh & 3;
  const int wid = threadIdx.x >> 5, lane = threadIdx.x & 31;
  const int i0  = blockIdx.x * 64 + wid * 16;
  const int row = lane & 15, hi = lane >> 4;

  const _Float16* QKb = QKt + (size_t)b * kN * 512;
  const _Float16* Kg  = QKb + kC + hh * kHD;                 // K[j][d] base
  const _Float16* Vg  = Vb + (size_t)b * kC * kN + (size_t)(hh * kHD) * kN;  // V[d][j]

  __shared__ __align__(16) _Float16 Ksh[2][64 * 72];
  __shared__ __align__(16) _Float16 Vsh[2][64 * 72];
  __shared__ __align__(16) _Float16 Psh[4][16 * 72];

  // Q A-frags: A(m=i,k=d) = qk_t[i0+m][hh*64+k]  -- contiguous
  v16h aq[2];
  #pragma unroll
  for (int s = 0; s < 2; ++s)
    aq[s] = load_a_contig(QKb + (size_t)(i0 + row) * 512 + hh * kHD + s * 32, hi);

  v8f accO[4] = {};
  float mrow[8], lrow[8];
  #pragma unroll
  for (int r = 0; r < 8; ++r) { mrow[r] = -1e30f; lrow[r] = 0.f; }

  // prologue: DMA chunk 0 into buffer 0
  if (threadIdx.x < 32) {
    tdm_load_tile_64x64((unsigned)(size_t)&Ksh[0][0], Kg, 512, kN);
    tdm_load_tile_64x64((unsigned)(size_t)&Vsh[0][0], Vg, kN, kC);
  }

  for (int jc = 0; jc < NCHUNK; ++jc) {
    const int buf = jc & 1;
    if (threadIdx.x < 32) __builtin_amdgcn_s_wait_tensorcnt(0);
    __syncthreads();   // chunk jc resident; everyone past chunk jc-1 compute
    if (jc + 1 < NCHUNK && threadIdx.x < 32) {
      const int j1 = (jc + 1) * JC;
      tdm_load_tile_64x64((unsigned)(size_t)&Ksh[buf ^ 1][0], Kg + (size_t)j1 * 512, 512, kN);
      tdm_load_tile_64x64((unsigned)(size_t)&Vsh[buf ^ 1][0], Vg + j1, kN, kC);
    }

    // S = Q K^T for 4 j-tiles: B(k=d,n=j) = Ksh[j][d]  (contiguous)
    v8f sacc[4];
    #pragma unroll
    for (int jt = 0; jt < 4; ++jt) {
      v8f s = {};
      #pragma unroll
      for (int ks = 0; ks < 2; ++ks) {
        v16h bf = load_b_contig(&Ksh[buf][(jt * 16 + row) * 72 + ks * 32 + hi * 16]);
        s = __builtin_amdgcn_wmma_f32_16x16x32_f16(false, aq[ks], false, bf,
                                                   (short)0, s, false, false);
      }
      sacc[jt] = s;
    }

    // online softmax over the 64-col chunk (row stats live at vgpr r, 16-lane group)
    float psum[8];
    #pragma unroll
    for (int r = 0; r < 8; ++r) {
      float m = fmaxf(fmaxf(sacc[0][r], sacc[1][r]), fmaxf(sacc[2][r], sacc[3][r]));
      #pragma unroll
      for (int off = 8; off >= 1; off >>= 1) m = fmaxf(m, __shfl_xor(m, off, 16));
      float mnew = fmaxf(mrow[r], m);
      float f = __expf(mrow[r] - mnew);
      mrow[r] = mnew;
      lrow[r] *= f;
      #pragma unroll
      for (int t = 0; t < 4; ++t) accO[t][r] *= f;
      psum[r] = 0.f;
    }

    // P = exp(S-m) -> per-wave LDS [i][j] (A-layout staging)
    #pragma unroll
    for (int jt = 0; jt < 4; ++jt)
      #pragma unroll
      for (int r = 0; r < 8; ++r) {
        float p = __expf(sacc[jt][r] - mrow[r]);
        psum[r] += p;
        Psh[wid][(r + 8 * hi) * 72 + jt * 16 + row] = (_Float16)p;
      }
    #pragma unroll
    for (int r = 0; r < 8; ++r) {
      float s = psum[r];
      #pragma unroll
      for (int off = 8; off >= 1; off >>= 1) s += __shfl_xor(s, off, 16);
      lrow[r] += s;
    }

    asm volatile("s_wait_dscnt 0" ::: "memory");  // same-wave cross-lane P handoff

    // O += P V^T : A(m=i,k=j)=Psh[i][j], B(k=j,n=d)=Vsh[d][j] (both contiguous)
    #pragma unroll
    for (int ks = 0; ks < 2; ++ks) {
      v16h ap = load_a_contig(&Psh[wid][row * 72 + ks * 32], hi);
      #pragma unroll
      for (int dt = 0; dt < 4; ++dt) {
        v16h bf = load_b_contig(&Vsh[buf][(dt * 16 + row) * 72 + ks * 32 + hi * 16]);
        accO[dt] = __builtin_amdgcn_wmma_f32_16x16x32_f16(false, ap, false, bf,
                                                          (short)0, accO[dt], false, false);
      }
    }
  }

  // normalize, write attn_t [s][c]
  _Float16* outp = At + (size_t)b * kN * kC;
  #pragma unroll
  for (int r = 0; r < 8; ++r) {
    float inv = 1.f / lrow[r];
    #pragma unroll
    for (int dt = 0; dt < 4; ++dt)
      outp[(size_t)(i0 + r + 8 * hi) * kC + hh * kHD + dt * 16 + row] =
          (_Float16)(accO[dt][r] * inv);
  }
}

// ---------------------------------------------------------------------------
// Kernel 5: proj GEMM + bias + residual (f32 out, [b][c][s] layout)
// Same 32x64 register-blocked wave tile as kernel 3.
// ---------------------------------------------------------------------------
__global__ void gemm_proj_kernel(const _Float16* __restrict__ W,    // [256,256]
                                 const _Float16* __restrict__ At,   // [B,1024,256]
                                 const float* __restrict__ x,
                                 const float* __restrict__ bias,
                                 float* __restrict__ out) {
  constexpr int K = 256;
  const int b  = blockIdx.z;
  const int m0 = blockIdx.x * 128;
  const int n0 = blockIdx.y * 128;
  const int wid = threadIdx.x >> 5, lane = threadIdx.x & 31;
  const int mw = m0 + (wid >> 1) * 32;
  const int nw = n0 + (wid & 1) * 64;
  const int row = lane & 15, hi = lane >> 4;
  const _Float16* Bp = At + (size_t)b * kN * kC;

  v8f acc[2][4] = {};
  for (int k0 = 0; k0 < K; k0 += 32) {
    v16h a0 = load_a_contig(W + (size_t)(mw + row) * K + k0, hi);
    v16h a1 = load_a_contig(W + (size_t)(mw + 16 + row) * K + k0, hi);
    #pragma unroll
    for (int t = 0; t < 4; ++t) {
      v16h bf = load_b_contig(Bp + (size_t)(nw + t * 16 + row) * kC + k0 + hi * 16);
      acc[0][t] = __builtin_amdgcn_wmma_f32_16x16x32_f16(false, a0, false, bf,
                                                         (short)0, acc[0][t], false, false);
      acc[1][t] = __builtin_amdgcn_wmma_f32_16x16x32_f16(false, a1, false, bf,
                                                         (short)0, acc[1][t], false, false);
    }
  }
  const float* xb = x + (size_t)b * kC * kN;
  float* ob = out + (size_t)b * kC * kN;
  #pragma unroll
  for (int mi = 0; mi < 2; ++mi)
    #pragma unroll
    for (int t = 0; t < 4; ++t)
      #pragma unroll
      for (int r = 0; r < 8; ++r) {
        const int mm = mw + mi * 16 + r + 8 * hi;
        const size_t idx = (size_t)mm * kN + nw + t * 16 + row;
        ob[idx] = xb[idx] + bias[mm] + acc[mi][t][r];
      }
}

// ---------------------------------------------------------------------------
extern "C" void kernel_launch(void* const* d_in, const int* in_sizes, int n_in,
                              void* d_out, int out_size, void* d_ws, size_t ws_size,
                              hipStream_t stream) {
  const float* x      = (const float*)d_in[0];
  const float* norm_w = (const float*)d_in[1];
  const float* norm_b = (const float*)d_in[2];
  const float* qkv_w  = (const float*)d_in[3];
  const float* proj_w = (const float*)d_in[4];
  const float* proj_b = (const float*)d_in[5];
  float* out = (float*)d_out;

  char* ws = (char*)d_ws;
  auto alloc = [&](size_t bytes) -> char* {
    char* p = ws;
    ws += (bytes + 255) & ~(size_t)255;
    return p;
  };
  _Float16* wq    = (_Float16*)alloc((size_t)768 * 256 * 2);
  _Float16* wp    = (_Float16*)alloc((size_t)256 * 256 * 2);
  _Float16* h_t   = (_Float16*)alloc((size_t)kB * kN * kC * 2);
  _Float16* qk_t  = (_Float16*)alloc((size_t)kB * kN * 512 * 2);
  _Float16* v_buf = (_Float16*)alloc((size_t)kB * kC * kN * 2);
  _Float16* at_t  = (_Float16*)alloc((size_t)kB * kN * kC * 2);

  // fold softmax scale (hd^-0.5 = 0.125) into Wq rows
  cvt_w_kernel<<<768, 256, 0, stream>>>(qkv_w, wq, 768 * 256, 256, 256, 0.125f);
  cvt_w_kernel<<<256, 256, 0, stream>>>(proj_w, wp, 256 * 256, 0, 256, 1.0f);
  groupnorm_kernel<<<kB * kG, 256, 0, stream>>>(x, norm_w, norm_b, h_t);
  gemm_qkv_kernel<<<dim3(6, 8, kB), 256, 0, stream>>>(wq, h_t, qk_t, v_buf);
  attn_kernel<<<dim3(16, 64), 128, 0, stream>>>(qk_t, v_buf, at_t);
  gemm_proj_kernel<<<dim3(2, 8, kB), 256, 0, stream>>>(wp, at_t, x, proj_b, out);
}
